// LanguageModel_18236431138929
// MI455X (gfx1250) — matrix operand order repaired and checked by reading, not compiled, not run on hardware
//
#include <hip/hip_runtime.h>

typedef unsigned short u16;
typedef __attribute__((ext_vector_type(16))) __bf16 bf16x16;
typedef __attribute__((ext_vector_type(8)))  __bf16 bf16x8;
typedef __attribute__((ext_vector_type(8)))  float  v8f;

// ---------------------------------------------------------------- helpers

__device__ inline u16 f2bf(float x) {
  union { float f; unsigned u; } v; v.f = x;
  unsigned u = v.u;
  u += 0x7fffu + ((u >> 16) & 1u);   // round-to-nearest-even
  return (u16)(u >> 16);
}

__device__ inline bf16x8 ld8(const u16* p) { return *(const bf16x8*)p; }

__device__ inline bf16x16 cat8(bf16x8 a, bf16x8 b) {
  return __builtin_shufflevector(a, b, 0,1,2,3,4,5,6,7,8,9,10,11,12,13,14,15);
}

__device__ inline v8f zero8() {
  v8f z;
#pragma unroll
  for (int i = 0; i < 8; ++i) z[i] = 0.f;
  return z;
}

__device__ inline float sigmoidf(float x) { return 1.f / (1.f + __expf(-x)); }

// Flat pointer into __shared__ -> raw LDS byte offset (aperture rule: LDS offset is the low 32 address bits).
__device__ inline unsigned lds_off(const void* p) {
  return (unsigned)(unsigned long long)p;
}

// gfx1250 async global->LDS copy, 16B per lane, tracked by ASYNCcnt (no VGPR round trip).
__device__ inline void async_cp_b128(unsigned lds_byte_off, const void* gptr) {
  asm volatile("global_load_async_to_lds_b128 %0, %1, off"
               :: "v"(lds_byte_off), "v"(gptr) : "memory");
}

// ---------------------------------------------------------------- generic bf16 WMMA GEMM
// C[M,N] = A[M,K](bf16) * B(bf16) + bias(+bias2), fused epilogue.
// transB==0: B is [N,K] row-major; transB==1: B is [K,N].
// flags: 1=ReLU, 2=gate-scaled accumulate, 4=also store bf16, 8=skip f32 store, 16=nontemporal f32 store.
// Block: 256 threads (8 waves), tile 128x128, K-step 32, double-buffered async LDS staging.

__global__ __launch_bounds__(256) void wmma_gemm_bf16(
    const u16* __restrict__ A, const u16* __restrict__ Bm,
    const float* __restrict__ bias, const float* __restrict__ bias2,
    float* __restrict__ C, u16* __restrict__ Cbf, const float* __restrict__ gatecol,
    int M, int N, int K, int transB, int flags)
{
  __shared__ uint4 Al4[2][128 * 32 / 8];
  __shared__ uint4 Bl4[2][128 * 32 / 8];

  const int tid  = threadIdx.x;
  const int lane = tid & 31;
  const int w    = tid >> 5;
  const int m0   = blockIdx.y * 128;
  const int n0   = blockIdx.x * 128;
  const int wm   = (w & 1) * 64;      // wave M offset in block tile
  const int wn   = (w >> 1) * 32;     // wave N offset in block tile
  const int ml   = lane & 15;
  const int hi   = lane >> 4;
  const int ak   = hi * 16;           // A-frag K offset (contiguous K, matches B)
  const int bk   = hi * 16;           // B-frag K offset

  v8f acc[4][2];
#pragma unroll
  for (int i = 0; i < 4; ++i)
#pragma unroll
    for (int j = 0; j < 2; ++j) acc[i][j] = zero8();

  const int arow = tid >> 1, ahalf = tid & 1;

  // ---- staging helpers (per k-tile, per buffer) ----
  auto stageA = [&](int k0, int buf) {            // async: 2 x b128 per thread
    const u16* s = A + (size_t)(m0 + arow) * K + k0 + ahalf * 16;
    unsigned d = lds_off((u16*)Al4[buf] + arow * 32 + ahalf * 16);
    async_cp_b128(d,      s);
    async_cp_b128(d + 16, s + 8);
  };
  auto stageB = [&](int k0, int buf) {
    if (transB) {                                 // [K,N] -> transpose (synchronous ds stores)
      u16* Bl = (u16*)Bl4[buf];
      for (int idx = tid; idx < 128 * 32; idx += 256) {
        int kk = idx >> 7, n = idx & 127;
        Bl[n * 32 + kk] = Bm[(size_t)(k0 + kk) * N + n0 + n];
      }
    } else {                                      // [N,K] -> async direct copy
      const u16* s = Bm + (size_t)(n0 + arow) * K + k0 + ahalf * 16;
      unsigned d = lds_off((u16*)Bl4[buf] + arow * 32 + ahalf * 16);
      async_cp_b128(d,      s);
      async_cp_b128(d + 16, s + 8);
    }
  };

  // ---- software pipeline: stage k+1 (buffer ^1) while computing k (buffer pb) ----
  stageA(0, 0);
  stageB(0, 0);
  int pb = 0;
  for (int k0 = 0; k0 < K; k0 += 32, pb ^= 1) {
    const bool next = (k0 + 32) < K;
    if (next) { stageA(k0 + 32, pb ^ 1); stageB(k0 + 32, pb ^ 1); }

    // wait for current buffer's async loads (in-order completion); leave next's in flight
    if (transB) {
      if (next) asm volatile("s_wait_asynccnt 0x2" ::: "memory");
      else      asm volatile("s_wait_asynccnt 0x0" ::: "memory");
    } else {
      if (next) asm volatile("s_wait_asynccnt 0x4" ::: "memory");
      else      asm volatile("s_wait_asynccnt 0x0" ::: "memory");
    }
    __syncthreads();

    const u16* Al = (const u16*)Al4[pb];
    const u16* Bl = (const u16*)Bl4[pb];
    bf16x16 af[4], bfg[2];
#pragma unroll
    for (int i = 0; i < 4; ++i) {
      const u16* p = Al + (wm + i * 16 + ml) * 32 + ak;
      af[i] = cat8(ld8(p), ld8(p + 8));
    }
#pragma unroll
    for (int j = 0; j < 2; ++j) {
      const u16* p = Bl + (wn + j * 16 + ml) * 32 + bk;
      bfg[j] = cat8(ld8(p), ld8(p + 8));
    }
#pragma unroll
    for (int i = 0; i < 4; ++i)
#pragma unroll
      for (int j = 0; j < 2; ++j)
        acc[i][j] = __builtin_amdgcn_wmma_f32_16x16x32_bf16(
            false, af[i], false, bfg[j], (short)0, acc[i][j], false, false);
    __syncthreads();   // next iteration's stage overwrites buffer pb
  }

  // epilogue (C/D layout: VGPR r -> M = r + (lane>=16 ? 8 : 0), N = lane&15)
#pragma unroll
  for (int j = 0; j < 2; ++j) {
    int nc = n0 + wn + j * 16 + ml;
    float bv = 0.f;
    if (bias)  bv += bias[nc];
    if (bias2) bv += bias2[nc];
#pragma unroll
    for (int i = 0; i < 4; ++i) {
      int mb = m0 + wm + i * 16 + hi * 8;
#pragma unroll
      for (int r = 0; r < 8; ++r) {
        int mr = mb + r;
        float v = acc[i][j][r] + bv;
        if (flags & 1) v = fmaxf(v, 0.f);
        size_t idx = (size_t)mr * N + nc;
        if (flags & 2) { v = C[idx] + gatecol[(size_t)mr * 8] * v; C[idx] = v; }
        else if (!(flags & 8)) {
          if (flags & 16) __builtin_nontemporal_store(v, &C[idx]);  // streaming logits
          else            C[idx] = v;
        }
        if (flags & 4) Cbf[idx] = f2bf(v);
      }
    }
  }
}

// ---------------------------------------------------------------- LSTM recurrence
// Single workgroup (1024 threads = 32 wave32 waves). pre: [4096,2048] (n = b*S+t).
// Per step: g = pre[:,t] + h @ Whh^T via WMMA (M=4 padded to 16, N=2048, K=512),
// then elementwise gate math in LDS. Writes out = h + resid (f32 + bf16).

__global__ __launch_bounds__(1024) void lstm_seq(
    const float* __restrict__ pre, const u16* __restrict__ whh,
    const float* __restrict__ resid, float* __restrict__ out,
    u16* __restrict__ outbf, int S)
{
  __shared__ uint4 hbf4[16 * 512 / 8];     // h as bf16, rows 4..15 stay zero (A pad)
  __shared__ float c_s[4 * 512];
  __shared__ float g_s[4 * 2048];
  u16* h_bf = (u16*)hbf4;

  const int tid  = threadIdx.x;
  const int lane = tid & 31;
  const int w    = tid >> 5;
  const int ml   = lane & 15;
  const int hi   = lane >> 4;
  const int ak   = hi * 16;
  const int bk   = hi * 16;

  for (int i = tid; i < 16 * 512; i += 1024) h_bf[i] = 0;
  for (int i = tid; i < 4 * 512; i += 1024) c_s[i] = 0.f;
  __syncthreads();

  for (int t = 0; t < S; ++t) {
    // ---- phase 1: g = h @ Whh^T (+pre) ; 128 N-tiles, 4 per wave
#pragma unroll
    for (int nn = 0; nn < 4; ++nn) {
      int nt = w * 4 + nn;                     // 0..127
      v8f acc = zero8();
      const u16* brow = whh + (size_t)(nt * 16 + ml) * 512;
#pragma unroll
      for (int kb = 0; kb < 512; kb += 32) {
        const u16* pa = h_bf + ml * 512 + kb + ak;
        bf16x16 a = cat8(ld8(pa), ld8(pa + 8));
        const u16* pb = brow + kb + bk;
        bf16x16 b = cat8(ld8(pb), ld8(pb + 8));
        acc = __builtin_amdgcn_wmma_f32_16x16x32_bf16(
            false, a, false, b, (short)0, acc, false, false);
      }
      int nc = nt * 16 + ml;
      if (lane < 16) {                         // rows M=0..3 valid (batch=4)
#pragma unroll
        for (int r = 0; r < 4; ++r)
          g_s[r * 2048 + nc] = acc[r] + pre[((size_t)r * S + t) * 2048 + nc];
      }
    }
    __syncthreads();

    // ---- phase 2: gate nonlinearity + state update (torch gate order i,f,g,o)
    for (int idx = tid; idx < 2048; idx += 1024) {
      int b = idx >> 9, j = idx & 511;
      const float* gb = &g_s[b * 2048];
      float iv = sigmoidf(gb[j]);
      float fv = sigmoidf(gb[512 + j]);
      float gv = tanhf(gb[1024 + j]);
      float ov = sigmoidf(gb[1536 + j]);
      float c  = fv * c_s[idx] + iv * gv;
      c_s[idx] = c;
      float h  = ov * tanhf(c);
      h_bf[b * 512 + j] = f2bf(h);
      size_t oidx = ((size_t)b * S + t) * 512 + j;
      float o = h + resid[oidx];
      out[oidx]  = o;
      outbf[oidx] = f2bf(o);
    }
    __syncthreads();
  }
}

// ---------------------------------------------------------------- MoE gating: top-4 softmax gates
__global__ void gate_topk(const float* __restrict__ x, const float* __restrict__ wg,
                          float* __restrict__ gates, int n, int H)
{
  int i = blockIdx.x * blockDim.x + threadIdx.x;
  if (i >= n) return;
  float lg[8];
#pragma unroll
  for (int e = 0; e < 8; ++e) lg[e] = 0.f;
  const float* xr = x + (size_t)i * H;
  for (int h = 0; h < H; ++h) {
    float xv = xr[h];
    const float* wr = wg + (size_t)h * 8;
#pragma unroll
    for (int e = 0; e < 8; ++e) lg[e] += xv * wr[e];
  }
  int ti[4]; float tv[4]; unsigned used = 0;
  for (int t = 0; t < 4; ++t) {
    int bi = 0; float bv = -3.0e38f;
    for (int e = 0; e < 8; ++e)
      if (!((used >> e) & 1u) && lg[e] > bv) { bv = lg[e]; bi = e; }
    used |= 1u << bi; ti[t] = bi; tv[t] = bv;
  }
  float mx = tv[0], ex[4], s = 0.f;
  for (int t = 0; t < 4; ++t) { ex[t] = __expf(tv[t] - mx); s += ex[t]; }
  float og[8];
#pragma unroll
  for (int e = 0; e < 8; ++e) og[e] = 0.f;
  for (int t = 0; t < 4; ++t) og[ti[t]] = ex[t] / s;
  float* gr = gates + (size_t)i * 8;
#pragma unroll
  for (int e = 0; e < 8; ++e) gr[e] = og[e];
}

// ---------------------------------------------------------------- aux loss (deterministic reduce)
__global__ void moe_aux(const float* __restrict__ gates, int n, float* __restrict__ outp)
{
  __shared__ float imp[8];
  __shared__ float ldc[8];
  int e = threadIdx.x;
  if (e < 8) {
    float s = 0.f, l = 0.f;
    for (int i = 0; i < n; ++i) {
      float g = gates[(size_t)i * 8 + e];
      s += g;
      if (g > 0.f) l += 1.f;
    }
    imp[e] = s; ldc[e] = l;
  }
  __syncthreads();
  if (e == 0) {
    float aux = 0.f;
    for (int k = 0; k < 2; ++k) {
      const float* v = k ? ldc : imp;
      float m = 0.f;
      for (int t = 0; t < 8; ++t) m += v[t];
      m *= 0.125f;
      float var = 0.f;
      for (int t = 0; t < 8; ++t) { float d = v[t] - m; var += d * d; }
      var *= (1.f / 7.f);                      // unbiased (ddof=1)
      aux += var / (m * m + 1e-10f);
    }
    outp[0] = aux * 0.01f;
  }
}

// ---------------------------------------------------------------- small elementwise kernels
__global__ void cvt_f32_bf16(const float* __restrict__ s, u16* __restrict__ d, int n)
{
  int i = blockIdx.x * blockDim.x + threadIdx.x;
  if (i < n) d[i] = f2bf(s[i]);
}

__global__ void zero_f32(float* __restrict__ p, int n)
{
  int i = blockIdx.x * blockDim.x + threadIdx.x;
  if (i < n) p[i] = 0.f;
}

__global__ void sigmoid_residual(const float* __restrict__ a, const float* __restrict__ r,
                                 float* __restrict__ o, u16* __restrict__ ob, int n)
{
  int i = blockIdx.x * blockDim.x + threadIdx.x;
  if (i < n) {
    float v = sigmoidf(a[i]) + r[i];
    o[i]  = v;
    ob[i] = f2bf(v);
  }
}

// ---------------------------------------------------------------- launcher
extern "C" void kernel_launch(void* const* d_in, const int* in_sizes, int n_in,
                              void* d_out, int out_size, void* d_ws, size_t ws_size,
                              hipStream_t stream)
{
  (void)in_sizes; (void)n_in; (void)ws_size;
  const int Ntok = 4096, H = 512, G = 2048, E = 8, V = 32000, S = 1024;

  const float* x     = (const float*)d_in[0];
  const float* We    = (const float*)d_in[1];
  const float* be    = (const float*)d_in[2];
  const float* Wih1  = (const float*)d_in[3];
  const float* Whh1  = (const float*)d_in[4];
  const float* bih1  = (const float*)d_in[5];
  const float* bhh1  = (const float*)d_in[6];
  const float* wgate = (const float*)d_in[7];
  const float* W1    = (const float*)d_in[8];
  const float* b1    = (const float*)d_in[9];
  const float* W2    = (const float*)d_in[10];
  const float* b2    = (const float*)d_in[11];
  const float* Wih2  = (const float*)d_in[12];
  const float* Whh2  = (const float*)d_in[13];
  const float* bih2  = (const float*)d_in[14];
  const float* bhh2  = (const float*)d_in[15];
  const float* Wfc   = (const float*)d_in[16];
  const float* bfc   = (const float*)d_in[17];

  char* ws = (char*)d_ws;
  size_t off = 0;
  auto alloc = [&](size_t bytes) -> char* {
    char* p = ws + off;
    off = (off + bytes + 255) & ~(size_t)255;
    return p;
  };
  u16*   xbf    = (u16*)alloc((size_t)Ntok * H * 2);
  u16*   WeB    = (u16*)alloc((size_t)H * H * 2);
  u16*   Wih1B  = (u16*)alloc((size_t)G * H * 2);
  u16*   Whh1B  = (u16*)alloc((size_t)G * H * 2);
  u16*   Wih2B  = (u16*)alloc((size_t)G * H * 2);
  u16*   Whh2B  = (u16*)alloc((size_t)G * H * 2);
  u16*   W1B    = (u16*)alloc((size_t)E * H * H * 2);
  u16*   W2B    = (u16*)alloc((size_t)E * H * H * 2);
  u16*   WfcB   = (u16*)alloc((size_t)V * H * 2);
  float* embedF = (float*)alloc((size_t)Ntok * H * 4);
  u16*   embedB = (u16*)alloc((size_t)Ntok * H * 2);
  float* preF   = (float*)alloc((size_t)Ntok * G * 4);
  float* out1F  = (float*)alloc((size_t)Ntok * H * 4);
  u16*   out1B  = (u16*)alloc((size_t)Ntok * H * 2);
  float* gates  = (float*)alloc((size_t)Ntok * E * 4);
  u16*   hbufB  = (u16*)alloc((size_t)Ntok * H * 2);
  float* moeAcc = (float*)alloc((size_t)Ntok * H * 4);
  float* moeF   = (float*)alloc((size_t)Ntok * H * 4);
  u16*   moeB   = (u16*)alloc((size_t)Ntok * H * 2);
  float* out2F  = (float*)alloc((size_t)Ntok * H * 4);
  u16*   out2B  = (u16*)alloc((size_t)Ntok * H * 2);

  auto cvt = [&](const float* s, u16* d, int n) {
    cvt_f32_bf16<<<(n + 255) / 256, 256, 0, stream>>>(s, d, n);
  };
  auto gemm = [&](const u16* A, const u16* Bm, const float* b1p, const float* b2p,
                  float* C, u16* Cbf, const float* gc, int M, int N, int K,
                  int tB, int flags) {
    dim3 grid(N / 128, M / 128);
    wmma_gemm_bf16<<<grid, 256, 0, stream>>>(A, Bm, b1p, b2p, C, Cbf, gc,
                                             M, N, K, tB, flags);
  };

  // weight + input conversion to bf16 (one pass; stays hot in L2 afterwards)
  cvt(x,    xbf,   Ntok * H);
  cvt(We,   WeB,   H * H);
  cvt(Wih1, Wih1B, G * H);
  cvt(Whh1, Whh1B, G * H);
  cvt(Wih2, Wih2B, G * H);
  cvt(Whh2, Whh2B, G * H);
  cvt(W1,   W1B,   E * H * H);
  cvt(W2,   W2B,   E * H * H);
  cvt(Wfc,  WfcB,  V * H);

  // embed = x @ We^T + be   (store f32 + bf16)
  gemm(xbf, WeB, be, nullptr, embedF, embedB, nullptr, Ntok, H, H, 0, 4);
  // pre1 = embed @ Wih1^T + bih1 + bhh1
  gemm(embedB, Wih1B, bih1, bhh1, preF, nullptr, nullptr, Ntok, G, H, 0, 0);
  // LSTM1 recurrence; out1 = lstm + embed
  lstm_seq<<<1, 1024, 0, stream>>>(preF, Whh1B, embedF, out1F, out1B, S);

  // MoE gating + aux loss
  gate_topk<<<Ntok / 256, 256, 0, stream>>>(out1F, wgate, gates, Ntok, H);
  moe_aux<<<1, 32, 0, stream>>>(gates, Ntok, (float*)d_out + (out_size - 1));

  // MoE experts: moeAcc = sum_e gate[:,e] * (relu(out1 @ W1[e] + b1[e]) @ W2[e] + b2[e])
  zero_f32<<<(Ntok * H + 255) / 256, 256, 0, stream>>>(moeAcc, Ntok * H);
  for (int e = 0; e < E; ++e) {
    gemm(out1B, W1B + (size_t)e * H * H, b1 + (size_t)e * H, nullptr,
         nullptr, hbufB, nullptr, Ntok, H, H, 1, 1 | 4 | 8);     // ReLU, bf16-only
    gemm(hbufB, W2B + (size_t)e * H * H, b2 + (size_t)e * H, nullptr,
         moeAcc, nullptr, gates + e, Ntok, H, H, 1, 2);          // gate-scaled accumulate
  }

  // moe_out = sigmoid(moeAcc) + out1
  sigmoid_residual<<<(Ntok * H + 255) / 256, 256, 0, stream>>>(moeAcc, out1F, moeF, moeB,
                                                               Ntok * H);
  // pre2 = moe_out @ Wih2^T + bih2 + bhh2
  gemm(moeB, Wih2B, bih2, bhh2, preF, nullptr, nullptr, Ntok, G, H, 0, 0);
  // LSTM2 recurrence; out2 = lstm + moe_out
  lstm_seq<<<1, 1024, 0, stream>>>(preF, Whh2B, moeF, out2F, out2B, S);

  // logits = out2 @ Wfc^T + bfc  -> d_out (dominant GEMM: 4096x32000x512, streaming NT store)
  gemm(out2B, WfcB, bfc, nullptr, (float*)d_out, nullptr, nullptr, Ntok, V, H, 0, 16);
}
